// LaneAwareProbabilityLearning_23347442221572
// MI455X (gfx1250) — compile-verified, matrix operand order checked by reading, not run on hardware
//
#include <hip/hip_runtime.h>
#include <cstdint>
#include <cstddef>

// Problem constants (match reference)
#define B_   128
#define L_   256
#define S_   256
#define H_   256
#define AD_  128
#define NL_  6
#define NB_  3
#define NH_  4
#define DH_  64
#define EPS_ 1e-5f
#define R_   (B_ * L_)   // 32768 rows for [B,L,*] flattened

typedef _Float16 half_t;
typedef __attribute__((ext_vector_type(16))) _Float16 v16h;
typedef __attribute__((ext_vector_type(8)))  _Float16 v8h;
typedef __attribute__((ext_vector_type(8)))  float    v8f;

__device__ __forceinline__ float silu_f(float v) { return v / (1.f + __expf(-v)); }

// Per-lane async global->LDS 16-byte copy (gfx1250 GLOBAL_LOAD_ASYNC_TO_LDS_B128,
// tracked by ASYNCcnt). ldsoff = wave-relative LDS byte offset (low 32 bits of the
// generic shared-aperture address), src = per-lane global address.
__device__ __forceinline__ void async_copy_b128(unsigned ldsoff, const void* src)
{
    asm volatile("global_load_async_to_lds_b128 %0, %1, off"
                 :: "v"(ldsoff), "v"(src) : "memory");
}
__device__ __forceinline__ void wait_async_zero()
{
    asm volatile("s_wait_asynccnt 0x0" ::: "memory");
}

// ---------------------------------------------------------------------------
// Generic WMMA GEMM:  C[z] = epilogue( A[z] (MxK) * B[z] (KxN) * scale + bias )
//   - fp16 inputs, fp32 WMMA accumulation (v_wmma_f32_16x16x32_f16), fp16 out
//   - block = 256 threads = 8 waves; tile 128(M) x 64(N) x 32(K)
//   - wave (wm,wn): wm in 0..3 covers 32 rows (2 A frags),
//                   wn in 0..1 covers 32 cols (2 B frags) -> 4 WMMA / K-step
//   - double-buffered LDS; tile i+1 staged with ASYNCcnt copies while tile i
//     is consumed by the matrix pipe (copy/compute overlap, 1 barrier/K-step)
//   - TB: B element (k,n) read as Bg[n*ldb + k] (pre-transposed / K^T case)
//   - two-level batch strides: z -> (zo = z/zdiv, zi = z%zdiv)
//   - HAS_ADD: += addp before activation (q_ssm);  HAS_RES: += resp after (mamba)
// ---------------------------------------------------------------------------
template<int ACT, bool TB, bool HAS_ADD, bool HAS_RES>
__global__ __launch_bounds__(256)
void gemm_wmma_kernel(const half_t* __restrict__ Ag, const half_t* __restrict__ Bg,
                      const float*  __restrict__ bias,
                      const half_t* __restrict__ addp, const half_t* __restrict__ resp,
                      half_t* __restrict__ Cg,
                      int M, int N, int K, int lda, int ldb, int ldc, int zdiv,
                      long long sAo, long long sAi, long long sBo, long long sBi,
                      long long sCo, long long sCi, float scale)
{
    __shared__ half_t At[2][128 * 40];  // A tiles, row-major [m][k], padded stride 40
    __shared__ half_t Bt[2][64 * 40];   // B tiles, transposed [n][k], padded stride 40

    const int t    = threadIdx.x;
    const int lane = t & 31;
    const int w    = t >> 5;
    const int wm   = w >> 1;          // 0..3 -> rows wm*32
    const int wn   = w & 1;           // 0..1 -> cols wn*32
    const int bm0  = blockIdx.y * 128;
    const int bn0  = blockIdx.x * 64;

    const int z  = blockIdx.z;
    const int zo = z / zdiv;
    const int zi = z - zo * zdiv;
    Ag += (size_t)zo * sAo + (size_t)zi * sAi;
    Bg += (size_t)zo * sBo + (size_t)zi * sBi;
    const size_t coff = (size_t)zo * sCo + (size_t)zi * sCi;
    Cg += coff;

    v8f acc00 = {}; v8f acc01 = {};
    v8f acc10 = {}; v8f acc11 = {};

    // cooperative-load thread mappings
    const int rB = t >> 2, sgB = t & 3;   // B (TB): 64 rows x (4 x 8 halves)
    const int kB = t >> 3, sg8 = t & 7;   // B non-TB: 32 k x (8 x 8 halves)

    // Stage one 128x32 A tile + 64x32 B tile (K-offset k0) into buffer `buf`.
    auto stage = [&](int k0, int buf) {
#pragma unroll
        for (int rep = 0; rep < 2; ++rep) {
            const int idx = t + rep * 256;
            const int r = idx >> 2, seg = idx & 3;
            const half_t* src = Ag + (size_t)(bm0 + r) * lda + k0 + seg * 8;
            async_copy_b128((unsigned)(uintptr_t)&At[buf][r * 40 + seg * 8], src);
        }
        if (TB) {
            const half_t* src = Bg + (size_t)(bn0 + rB) * ldb + k0 + sgB * 8;
            async_copy_b128((unsigned)(uintptr_t)&Bt[buf][rB * 40 + sgB * 8], src);
        } else {
            // needs an element transpose -> through registers + ds stores
            const half_t* src = Bg + (size_t)(k0 + kB) * ldb + bn0 + sg8 * 8;
            v8h tmp = *(const v8h*)src;
#pragma unroll
            for (int i = 0; i < 8; ++i) Bt[buf][(sg8 * 8 + i) * 40 + kB] = tmp[i];
        }
    };

    const int nk = K >> 5;               // K / 32 (all K are multiples of 32)

    // prologue: stage tile 0
    stage(0, 0);
    wait_async_zero();                   // hand-issued asyncs aren't barrier-tracked
    __syncthreads();

    for (int ki = 0; ki < nk; ++ki) {
        const int cur = ki & 1;
        // kick off next tile's copies into the other buffer (overlaps with WMMA)
        if (ki + 1 < nk) {
            stage((ki + 1) << 5, cur ^ 1);
            // warm caches for the tile after next while we're at it
            if (ki + 2 < nk)
                __builtin_prefetch(Ag + (size_t)(bm0 + (t >> 2)) * lda +
                                   ((ki + 2) << 5) + (t & 3) * 8, 0, 3);
        }

        // ---- build fragments per CDNA5 16-bit A/B layouts ----
        // lane l: M(or N) = l&15 ; K groups: lanes<16 -> K {0..7,16..23},
        //                                    lanes>=16 -> K {8..15,24..31}
        const half_t* Ab = At[cur];
        const half_t* Bb = Bt[cur];
        const int kb = (lane >> 4) * 8;
        union Frag { v16h v; v8h p[2]; };
        Frag fa0, fa1, fb0, fb1;
        {
            const int r0 = (wm * 32 + (lane & 15)) * 40;
            fa0.p[0] = *(const v8h*)&Ab[r0 + kb];
            fa0.p[1] = *(const v8h*)&Ab[r0 + 16 + kb];
            const int r1 = r0 + 16 * 40;
            fa1.p[0] = *(const v8h*)&Ab[r1 + kb];
            fa1.p[1] = *(const v8h*)&Ab[r1 + 16 + kb];
            const int n0 = (wn * 32 + (lane & 15)) * 40;
            fb0.p[0] = *(const v8h*)&Bb[n0 + kb];
            fb0.p[1] = *(const v8h*)&Bb[n0 + 16 + kb];
            const int n1 = n0 + 16 * 40;
            fb1.p[0] = *(const v8h*)&Bb[n1 + kb];
            fb1.p[1] = *(const v8h*)&Bb[n1 + 16 + kb];
        }
        acc00 = __builtin_amdgcn_wmma_f32_16x16x32_f16(false, fa0.v, false, fb0.v,
                                                       (short)0, acc00, false, false);
        acc01 = __builtin_amdgcn_wmma_f32_16x16x32_f16(false, fa0.v, false, fb1.v,
                                                       (short)0, acc01, false, false);
        acc10 = __builtin_amdgcn_wmma_f32_16x16x32_f16(false, fa1.v, false, fb0.v,
                                                       (short)0, acc10, false, false);
        acc11 = __builtin_amdgcn_wmma_f32_16x16x32_f16(false, fa1.v, false, fb1.v,
                                                       (short)0, acc11, false, false);

        // drain this wave's in-flight asyncs, then block-wide sync: after this,
        // buffer cur^1 is fully written and buffer cur is safe to overwrite next.
        wait_async_zero();
        __syncthreads();
    }

    // ---- epilogue: C/D layout: lane l holds col N=l&15; VGPR i -> row i+8*(l>>4)
    const int colL = lane & 15;
    const int rsel = (lane >> 4) * 8;
    auto epi = [&](v8f a, int mi, int nn) {
        const int gN = bn0 + wn * 32 + nn * 16 + colL;
        const float bv = bias ? bias[gN] : 0.f;
#pragma unroll
        for (int i = 0; i < 8; ++i) {
            const int gM = bm0 + wm * 32 + mi * 16 + rsel + i;
            const size_t idx = (size_t)gM * ldc + gN;
            float v = a[i] * scale + bv;
            if (HAS_ADD) v += (float)addp[coff + idx];
            if (ACT == 1) v = silu_f(v);
            if (HAS_RES) v += (float)resp[coff + idx];
            Cg[idx] = (half_t)v;
        }
    };
    epi(acc00, 0, 0); epi(acc01, 0, 1);
    epi(acc10, 1, 0); epi(acc11, 1, 1);
}

// ---------------------------------------------------------------------------
// Elementwise / normalization kernels
// ---------------------------------------------------------------------------
__global__ __launch_bounds__(256)
void cvt_f2h(const float* __restrict__ in, half_t* __restrict__ out, int n)
{
    int i = blockIdx.x * 256 + threadIdx.x;
    if (i < n) out[i] = (half_t)in[i];
}

// in: [K,N] row-major fp32 -> out: [N,K] fp16 (for TRANS_B weight path)
__global__ __launch_bounds__(256)
void cvt_f2h_T(const float* __restrict__ in, half_t* __restrict__ out, int K, int N)
{
    int i = blockIdx.x * 256 + threadIdx.x;
    if (i < K * N) {
        int k = i / N, n = i - k * N;
        out[(size_t)n * K + k] = (half_t)in[i];
    }
}

// out[i,k] = in[i,k] * softplus(delta[k])   (fused A_d / B_d build)
__global__ __launch_bounds__(256)
void scale_cols_f2h(const float* __restrict__ in, const float* __restrict__ delta,
                    half_t* __restrict__ out, int n, int N)
{
    int i = blockIdx.x * 256 + threadIdx.x;
    if (i < n) {
        int k = i % N;
        float d  = delta[k];
        float sp = (d > 20.f) ? d : log1pf(__expf(d));
        out[i] = (half_t)(in[i] * sp);
    }
}

// per-row LayerNorm over H_=256 with gain/beta (mamba blocks)
__global__ __launch_bounds__(256)
void layernorm256(const half_t* __restrict__ x, const float* __restrict__ g,
                  const float* __restrict__ be, half_t* __restrict__ out)
{
    __shared__ float red[256];
    const size_t row = blockIdx.x;
    const int t = threadIdx.x;
    float v = (float)x[row * 256 + t];
    red[t] = v; __syncthreads();
    for (int s = 128; s > 0; s >>= 1) { if (t < s) red[t] += red[t + s]; __syncthreads(); }
    float mu = red[0] * (1.f / 256.f);
    __syncthreads();
    float d = v - mu;
    red[t] = d * d; __syncthreads();
    for (int s = 128; s > 0; s >>= 1) { if (t < s) red[t] += red[t + s]; __syncthreads(); }
    float var = red[0] * (1.f / 256.f);
    out[row * 256 + t] = (half_t)(d * rsqrtf(var + EPS_) * g[t] + be[t]);
}

// in-place softmax over rows of length 256 (attention scores)
__global__ __launch_bounds__(256)
void softmax256(half_t* __restrict__ x)
{
    __shared__ float red[256];
    const size_t row = blockIdx.x;
    const int t = threadIdx.x;
    float v = (float)x[row * 256 + t];
    red[t] = v; __syncthreads();
    for (int s = 128; s > 0; s >>= 1) { if (t < s) red[t] = fmaxf(red[t], red[t + s]); __syncthreads(); }
    float mx = red[0];
    __syncthreads();
    float e = __expf(v - mx);
    red[t] = e; __syncthreads();
    for (int s = 128; s > 0; s >>= 1) { if (t < s) red[t] += red[t + s]; __syncthreads(); }
    x[row * 256 + t] = (half_t)(e / red[0]);
}

// InstanceNorm1d: normalize over seq dim L per (batch, channel)
__global__ __launch_bounds__(256)
void instnorm_seq(const half_t* __restrict__ x, half_t* __restrict__ out)
{
    __shared__ float red[256];
    const int b = blockIdx.x >> 8;      // / H_
    const int c = blockIdx.x & 255;     // % H_
    const int t = threadIdx.x;          // seq position
    const size_t idx = ((size_t)b * L_ + t) * H_ + c;
    float v = (float)x[idx];
    red[t] = v; __syncthreads();
    for (int s = 128; s > 0; s >>= 1) { if (t < s) red[t] += red[t + s]; __syncthreads(); }
    float mu = red[0] * (1.f / 256.f);
    __syncthreads();
    float d = v - mu;
    red[t] = d * d; __syncthreads();
    for (int s = 128; s > 0; s >>= 1) { if (t < s) red[t] += red[t + s]; __syncthreads(); }
    float var = red[0] * (1.f / 256.f);
    out[idx] = (half_t)(d * rsqrtf(var + EPS_));
}

// logits = x @ Wmlp + bmlp ; softmax over NL_=6 ; argmax  (N too small for WMMA)
__global__ __launch_bounds__(256)
void head_kernel(const half_t* __restrict__ x, const float* __restrict__ W,
                 const float* __restrict__ b, float* __restrict__ probs,
                 int* __restrict__ preds)
{
    const int row = blockIdx.x * 256 + threadIdx.x;   // < R_
    float acc[NL_];
#pragma unroll
    for (int j = 0; j < NL_; ++j) acc[j] = b[j];
    const half_t* xr = x + (size_t)row * H_;
    for (int k = 0; k < H_; ++k) {
        float xv = (float)xr[k];
#pragma unroll
        for (int j = 0; j < NL_; ++j) acc[j] += xv * W[k * NL_ + j];
    }
    float mx = acc[0];
#pragma unroll
    for (int j = 1; j < NL_; ++j) mx = fmaxf(mx, acc[j]);
    float s = 0.f;
#pragma unroll
    for (int j = 0; j < NL_; ++j) { acc[j] = __expf(acc[j] - mx); s += acc[j]; }
    float inv = 1.f / s;
    int arg = 0; float best = -1.f;
#pragma unroll
    for (int j = 0; j < NL_; ++j) {
        float p = acc[j] * inv;
        probs[(size_t)row * NL_ + j] = p;
        if (p > best) { best = p; arg = j; }
    }
    preds[row] = arg;
}

// ---------------------------------------------------------------------------
// Host-side GEMM dispatch
// ---------------------------------------------------------------------------
struct GP {
    const half_t *A, *B; const float* bias; const half_t *add, *res; half_t* C;
    int M, N, K, lda, ldb, ldc, zdiv, batch;
    long long sAo, sAi, sBo, sBi, sCo, sCi; float scale;
};

template<int ACT, bool TB, bool ADD, bool RES>
static void run_gemm(const GP& g, hipStream_t st)
{
    dim3 grid(g.N / 64, g.M / 128, g.batch);
    gemm_wmma_kernel<ACT, TB, ADD, RES><<<grid, 256, 0, st>>>(
        g.A, g.B, g.bias, g.add, g.res, g.C, g.M, g.N, g.K, g.lda, g.ldb, g.ldc,
        g.zdiv, g.sAo, g.sAi, g.sBo, g.sBi, g.sCo, g.sCi, g.scale);
}

static GP gp_basic(const half_t* A, const half_t* B, const float* bias, half_t* C,
                   int M, int N, int K, int lda, int ldb, int ldc)
{
    GP g{}; g.A = A; g.B = B; g.bias = bias; g.add = nullptr; g.res = nullptr; g.C = C;
    g.M = M; g.N = N; g.K = K; g.lda = lda; g.ldb = ldb; g.ldc = ldc;
    g.zdiv = 1; g.batch = 1; g.sAo = g.sAi = g.sBo = g.sBi = g.sCo = g.sCi = 0;
    g.scale = 1.f; return g;
}

// ---------------------------------------------------------------------------
extern "C" void kernel_launch(void* const* d_in, const int* in_sizes, int n_in,
                              void* d_out, int out_size, void* d_ws, size_t ws_size,
                              hipStream_t stream)
{
    const float* hlf   = (const float*)d_in[0];
    const float* lane  = (const float*)d_in[1];
    const float* Wm    = (const float*)d_in[2];
    const float* bm    = (const float*)d_in[3];
    const float* Wn    = (const float*)d_in[4];
    const float* bn    = (const float*)d_in[5];
    const float* Wc    = (const float*)d_in[6];
    const float* bc    = (const float*)d_in[7];
    const float* Amat  = (const float*)d_in[8];
    const float* Bmat  = (const float*)d_in[9];
    const float* delta = (const float*)d_in[10];
    const float* mbW   = (const float*)d_in[11];
    const float* mbb   = (const float*)d_in[12];
    const float* mbg   = (const float*)d_in[13];
    const float* mbbe  = (const float*)d_in[14];
    const float* Wqkv  = (const float*)d_in[15];
    const float* bqkv  = (const float*)d_in[16];
    const float* Wo    = (const float*)d_in[17];
    const float* bo    = (const float*)d_in[18];
    const float* Wmlp  = (const float*)d_in[19];
    const float* bmlp  = (const float*)d_in[20];

    // ---- workspace carve-out (fp16, 256B aligned), with lifetime-based reuse
    size_t cur = 0;
    char* base = (char*)d_ws;
    auto nexth = [&](size_t halves) -> half_t* {
        half_t* p = (half_t*)(base + cur);
        cur += ((halves * 2 + 255) & ~(size_t)255);
        return p;
    };
    half_t* x16   = nexth((size_t)R_ * AD_);        // lane_inputs fp16
    half_t* hlf16 = nexth((size_t)R_ * H_);         // hlf fp16  -> later q_final
    half_t* m16   = nexth((size_t)R_ * H_);         // m         -> later q_ssm
    half_t* n16   = nexth((size_t)R_ * H_);         // n         -> later ctx
    half_t* q16   = nexth((size_t)R_ * H_);         // m' / mamba x -> later Q
    half_t* ln16  = nexth((size_t)R_ * H_);         // LN scratch -> tmp -> cross
    half_t* KV16  = nexth((size_t)R_ * 2 * H_);     // [B,S,512] = [K|V]
    half_t* at16  = nexth((size_t)B_ * NH_ * L_ * S_); // scores/attn
    half_t* WmT   = nexth(H_ * AD_);
    half_t* WnT   = nexth(H_ * AD_);
    half_t* Wc16  = nexth(H_ * H_);                 // Wc rows are already [n][k]
    half_t* mbWT  = nexth((size_t)NB_ * H_ * H_);
    half_t* WqkvT = nexth((size_t)3 * H_ * H_);
    half_t* WoT   = nexth(H_ * H_);
    half_t* Ad16  = nexth(H_ * H_);
    half_t* Bd16  = nexth(H_ * H_);

    auto blocks = [](int n) { return (n + 255) / 256; };

    // ---- convert inputs & weights to fp16
    cvt_f2h<<<blocks(R_ * AD_), 256, 0, stream>>>(lane, x16, R_ * AD_);
    cvt_f2h<<<blocks(R_ * H_), 256, 0, stream>>>(hlf, hlf16, R_ * H_);
    cvt_f2h_T<<<blocks(AD_ * H_), 256, 0, stream>>>(Wm, WmT, AD_, H_);
    cvt_f2h_T<<<blocks(AD_ * H_), 256, 0, stream>>>(Wn, WnT, AD_, H_);
    cvt_f2h<<<blocks(H_ * H_), 256, 0, stream>>>(Wc, Wc16, H_ * H_);  // m@Wc.T: B[k][n]=Wc[n][k]
    for (int i = 0; i < NB_; ++i)
        cvt_f2h_T<<<blocks(H_ * H_), 256, 0, stream>>>(mbW + (size_t)i * H_ * H_,
                                                       mbWT + (size_t)i * H_ * H_, H_, H_);
    cvt_f2h_T<<<blocks(H_ * 3 * H_), 256, 0, stream>>>(Wqkv, WqkvT, H_, 3 * H_);
    cvt_f2h_T<<<blocks(H_ * H_), 256, 0, stream>>>(Wo, WoT, H_, H_);
    scale_cols_f2h<<<blocks(H_ * H_), 256, 0, stream>>>(Amat, delta, Ad16, H_ * H_, H_);
    scale_cols_f2h<<<blocks(H_ * H_), 256, 0, stream>>>(Bmat, delta, Bd16, H_ * H_, H_);

    // ---- 1,2: m = x@Wm + bm ; n = x@Wn + bn
    { GP g = gp_basic(x16, WmT, bm, m16, R_, H_, AD_, AD_, AD_, H_); run_gemm<0, true, false, false>(g, stream); }
    { GP g = gp_basic(x16, WnT, bn, n16, R_, H_, AD_, AD_, AD_, H_); run_gemm<0, true, false, false>(g, stream); }

    // ---- 3: m' = silu(m @ Wc.T + bc)
    { GP g = gp_basic(m16, Wc16, bc, q16, R_, H_, H_, H_, H_, H_); run_gemm<1, true, false, false>(g, stream); }

    // ---- 4: mamba blocks: x = x + silu(LN(x) @ W + b), in-place on q16
    for (int i = 0; i < NB_; ++i) {
        layernorm256<<<R_, 256, 0, stream>>>(q16, mbg + i * H_, mbbe + i * H_, ln16);
        GP g = gp_basic(ln16, mbWT + (size_t)i * H_ * H_, mbb + i * H_, q16, R_, H_, H_, H_, H_, H_);
        g.res = q16;
        run_gemm<1, true, false, true>(g, stream);
    }

    // ---- 5: tmp = A_d @ q   (batched over B, shared A)
    {
        GP g = gp_basic(Ad16, q16, nullptr, ln16, L_, H_, H_, H_, H_, H_);
        g.batch = B_; g.sBo = (long long)L_ * H_; g.sCo = (long long)L_ * H_;
        run_gemm<0, false, false, false>(g, stream);
    }
    // ---- 6: q_ssm = silu(tmp + B_d @ n)  -> m16
    {
        GP g = gp_basic(Bd16, n16, nullptr, m16, L_, H_, H_, H_, H_, H_);
        g.batch = B_; g.sBo = (long long)L_ * H_; g.sCo = (long long)L_ * H_;
        g.add = ln16;
        run_gemm<1, false, true, false>(g, stream);
    }

    // ---- 7: Q = q_ssm @ Wq + bq  -> q16
    { GP g = gp_basic(m16, WqkvT, bqkv, q16, R_, H_, H_, H_, H_, H_); run_gemm<0, true, false, false>(g, stream); }
    // ---- 8: [K|V] = hlf @ W(kv) + b(kv)  -> KV16 (ldc = 512)
    { GP g = gp_basic(hlf16, WqkvT + (size_t)H_ * H_, bqkv + H_, KV16, R_, 2 * H_, H_, H_, H_, 2 * H_);
      run_gemm<0, true, false, false>(g, stream); }

    // ---- 9: scores = (Q_bh @ K_bh^T) / sqrt(dh)  -> at16, batch z = b*NH+h
    {
        GP g = gp_basic(q16, KV16, nullptr, at16, L_, S_, DH_, H_, 2 * H_, S_);
        g.batch = B_ * NH_; g.zdiv = NH_;
        g.sAo = (long long)L_ * H_;      g.sAi = DH_;
        g.sBo = (long long)S_ * 2 * H_;  g.sBi = DH_;
        g.sCo = (long long)NH_ * L_ * S_; g.sCi = (long long)L_ * S_;
        g.scale = 0.125f;                 // 1/sqrt(64)
        run_gemm<0, true, false, false>(g, stream);
    }
    // ---- 10: softmax over S (in-place)
    softmax256<<<B_ * NH_ * L_, 256, 0, stream>>>(at16);

    // ---- 11: ctx = attn @ V_bh  -> n16 (column block h*64)
    {
        GP g = gp_basic(at16, KV16 + H_, nullptr, n16, L_, DH_, S_, S_, 2 * H_, H_);
        g.batch = B_ * NH_; g.zdiv = NH_;
        g.sAo = (long long)NH_ * L_ * S_; g.sAi = (long long)L_ * S_;
        g.sBo = (long long)S_ * 2 * H_;   g.sBi = DH_;
        g.sCo = (long long)L_ * H_;       g.sCi = DH_;
        run_gemm<0, false, false, false>(g, stream);
    }

    // ---- 12: cross = ctx @ Wo + bo  -> ln16
    { GP g = gp_basic(n16, WoT, bo, ln16, R_, H_, H_, H_, H_, H_); run_gemm<0, true, false, false>(g, stream); }

    // ---- 13: InstanceNorm over seq dim -> hlf16 (reused as q_final)
    instnorm_seq<<<B_ * H_, 256, 0, stream>>>(ln16, hlf16);

    // ---- 14: head: logits/softmax/argmax -> d_out (probs fp32, then preds i32)
    float* probs = (float*)d_out;
    int*   preds = (int*)(probs + (size_t)R_ * NL_);
    head_kernel<<<R_ / 256, 256, 0, stream>>>(hlf16, Wmlp, bmlp, probs, preds);

    (void)in_sizes; (void)n_in; (void)out_size; (void)ws_size;
}
